// CIM_SM_Linear_103079215222
// MI455X (gfx1250) — 8-bit matrix operand order repaired and checked by reading, compiled, not run on hardware
//
#include <hip/hip_runtime.h>
#include <stdint.h>


typedef __attribute__((ext_vector_type(8))) int v8i;
typedef __attribute__((ext_vector_type(4))) int v4i;

#define IN_F     1024
#define OUT_F    1024
#define T_TOKENS 2048
#define CIM_EPS  1e-8f

__global__ void quant_rows_kernel(const float* __restrict__ src,
                                  int8_t* __restrict__ q,
                                  float* __restrict__ scale,
                                  float qlo, float qhi)
{
    __shared__ float red[256];
    const int row = blockIdx.x;
    const int tid = threadIdx.x;
    const float* s = src + (size_t)row * IN_F;

    float m = 0.0f;
    for (int i = tid; i < IN_F; i += 256) m = fmaxf(m, fabsf(s[i]));
    red[tid] = m;
    __syncthreads();
    for (int off = 128; off > 0; off >>= 1) {
        if (tid < off) red[tid] = fmaxf(red[tid], red[tid + off]);
        __syncthreads();
    }
    const float sc = red[0] * (1.0f / 127.0f);
    if (tid == 0) scale[row] = sc;

    const float denom = sc + CIM_EPS;
    int8_t* qr = q + (size_t)row * IN_F;
    for (int i = tid; i < IN_F; i += 256) {
        float v = rintf(s[i] / denom);
        v = fminf(fmaxf(v, qlo), qhi);
        qr[i] = (int8_t)v;
    }
}

__global__ void cim_gemm_kernel(const int8_t* __restrict__ xq,
                                const int8_t* __restrict__ wq,
                                const float* __restrict__ sx,
                                const float* __restrict__ sw,
                                const float* __restrict__ bias,
                                float* __restrict__ out)
{
    const int lane   = threadIdx.x;
    const int wave   = threadIdx.y;
    const int tile_n = blockIdx.x;
    const int tile_m = blockIdx.y * 8 + wave;
    const int half   = lane >> 4;
    const int sub    = lane & 15;

    const int8_t* arow = xq + (size_t)(tile_m * 16 + sub) * IN_F + half * 8;
    const int8_t* brow = wq + (size_t)(tile_n * 16 + sub) * IN_F + half * 8;

    v8i acc = {0, 0, 0, 0, 0, 0, 0, 0};

#pragma unroll
    for (int k = 0; k < IN_F; k += 64) {
        union { v8i v; uint64_t d[4]; } a;
        const uint64_t* ap = (const uint64_t*)(arow + k);
        a.d[0] = ap[0];
        a.d[1] = ap[2];
        a.d[2] = ap[4];
        a.d[3] = ap[6];

        union { v8i v; uint64_t d[4]; } b;
        const uint64_t* bp = (const uint64_t*)(brow + k);
        b.d[0] = bp[0];
        b.d[1] = bp[2];
        b.d[2] = bp[4];
        b.d[3] = bp[6];

        acc = __builtin_amdgcn_wmma_i32_16x16x64_iu8(
            true, a.v, true, b.v, acc, false, false);
    }

    const int   o   = tile_n * 16 + sub;
    const float swo = sw[o];
    const float bo  = bias[o];
#pragma unroll
    for (int j = 0; j < 8; ++j) {
        const int t = tile_m * 16 + half * 8 + j;
        out[(size_t)t * OUT_F + o] = (float)acc[j] * sx[t] * swo + bo;
    }
}

extern "C" void kernel_launch(void* const* d_in, const int* in_sizes, int n_in,
                              void* d_out, int out_size, void* d_ws, size_t ws_size,
                              hipStream_t stream) {
    const float* x    = (const float*)d_in[0];
    const float* w    = (const float*)d_in[1];
    const float* bias = (const float*)d_in[2];
    float* out = (float*)d_out;

    uint8_t* ws = (uint8_t*)d_ws;
    int8_t* xq = (int8_t*)ws;
    int8_t* wq = (int8_t*)(ws + (size_t)T_TOKENS * IN_F);
    float*  sx = (float*)(ws + (size_t)(T_TOKENS + OUT_F) * IN_F);
    float*  sw = sx + T_TOKENS;

    quant_rows_kernel<<<T_TOKENS, 256, 0, stream>>>(x, xq, sx, -128.0f, 127.0f);
    quant_rows_kernel<<<OUT_F, 256, 0, stream>>>(w, wq, sw, -127.0f, 127.0f);

    dim3 grid(OUT_F / 16, T_TOKENS / 16 / 8);
    dim3 block(32, 8);
    cim_gemm_kernel<<<grid, block, 0, stream>>>(xq, wq, sx, sw, bias, out);
}
